// GraphGRU_11656541241782
// MI455X (gfx1250) — matrix operand order repaired and checked by reading, not compiled, not run on hardware
//
#include <hip/hip_runtime.h>

// GraphGRU on MI455X (gfx1250): bf16 WMMA (V_WMMA_F32_16X16X32_BF16).
// 16 messages/block = 256 gathered neighbor rows staged in the 320KB WGP LDS,
// both as f32 (exact gating/sums) and bf16 (WMMA A-fragments, converted ONCE).
// 256-thread blocks = 8 wave32 waves, one 16-col N-tile per wave.

#define HID 128
#define NEI 16
#define LDH  132  // f32 LDS row stride (floats): 132*4B -> 4-bank rotation/row
#define LDBH 136  // bf16 LDS row stride (shorts): 136*2B -> 4-bank rotation/row

typedef __attribute__((ext_vector_type(16))) __bf16 v16bf;
typedef __attribute__((ext_vector_type(8)))  float  v8f;

union BFvec { v16bf v; unsigned short s[16]; unsigned int u[8]; };

__device__ __forceinline__ unsigned short f2bf(float f) {
  unsigned u = __builtin_bit_cast(unsigned, f);
  u += 0x7FFFu + ((u >> 16) & 1u);   // round-to-nearest-even
  return (unsigned short)(u >> 16);
}
__device__ __forceinline__ unsigned pk2(float a, float b) {
  return (unsigned)f2bf(a) | ((unsigned)f2bf(b) << 16);
}

// B fragment (32x16 bf16, K x N): lane -> column lane%16, 16 contiguous K at kbase.
__device__ __forceinline__ v16bf load_bfrag(const unsigned short* __restrict__ Wrow, int kbase) {
  BFvec r;
  const uint4* p = (const uint4*)(Wrow + kbase);   // 32B contiguous
  *(uint4*)(&r.u[0]) = p[0];
  *(uint4*)(&r.u[4]) = p[1];
  return r.v;
}

// A fragment (16x32 bf16, M x K) from a bf16 LDS row: 16 contiguous K [k0,k0+16).
__device__ __forceinline__ v16bf load_afrag(const unsigned short* row, int k0) {
  BFvec r;
  *(uint4*)(&r.u[0]) = *(const uint4*)(row + k0);        // ds_load_b128
  *(uint4*)(&r.u[4]) = *(const uint4*)(row + k0 + 8);    // ds_load_b128
  return r.v;
}

__device__ __forceinline__ v8f wmma_bf16(v16bf a, v16bf b, v8f c) {
  return __builtin_amdgcn_wmma_f32_16x16x32_bf16(false, a, false, b, (short)0, c, false, false);
}

__device__ __forceinline__ float sigmoidf_(float v) { return 1.0f / (1.0f + __expf(-v)); }

// One-time f32 -> bf16 weight conversion into workspace:
// layout: Wz[128x256] @0, Wr[128x128] @32768, Ur[128x128] @49152, Wh[128x256] @65536
__global__ void gru_cvt_weights(const float* __restrict__ Wz, const float* __restrict__ Wr,
                                const float* __restrict__ Ur, const float* __restrict__ Wh,
                                unsigned short* __restrict__ out) {
  int i = blockIdx.x * blockDim.x + threadIdx.x;
  if (i < 32768) out[i]          = f2bf(Wz[i]);
  if (i < 16384) out[32768 + i]  = f2bf(Wr[i]);
  if (i < 16384) out[49152 + i]  = f2bf(Ur[i]);
  if (i < 32768) out[65536 + i]  = f2bf(Wh[i]);
}

__global__ __launch_bounds__(256)
void gru_depth(const float* __restrict__ h_in,
               const float* __restrict__ x,
               const int*   __restrict__ graph,
               const unsigned short* __restrict__ Wbf,
               const float* __restrict__ Wz_b,
               const float* __restrict__ Ur_b,
               const float* __restrict__ Wh_b,
               float* __restrict__ h_out)
{
  extern __shared__ float smem[];
  float* sh_hnei = smem;                                   // [256][LDH] f32 neighbors
  float* sh_r1   = sh_hnei + 256 * LDH;                    // [16][LDH]  r1 / h_out staging
  float* sh_sumh = sh_r1   + 16  * LDH;                    // [16][LDH]  f32 sum_h
  unsigned short* sh_hnei_bf = (unsigned short*)(sh_sumh + 16 * LDH); // [256][LDBH]
  unsigned short* sh_x_bf    = sh_hnei_bf + 256 * LDBH;    // [16][LDBH]
  unsigned short* sh_sumh_bf = sh_x_bf    + 16  * LDBH;    // [16][LDBH]
  unsigned short* sh_sumg_bf = sh_sumh_bf + 16  * LDBH;    // [16][LDBH]

  const unsigned short* WzB = Wbf;          // 128 x 256
  const unsigned short* WrB = Wbf + 32768;  // 128 x 128
  const unsigned short* UrB = Wbf + 49152;  // 128 x 128
  const unsigned short* WhB = Wbf + 65536;  // 128 x 256

  const int tid  = threadIdx.x;
  const int msg0 = blockIdx.x * 16;
  const int wv   = tid >> 5;        // wave id = N-tile index (0..7)
  const int lane = tid & 31;
  const int n_l  = lane & 15;       // column / A-row within tile
  const int half = lane >> 4;       // lane half select
  const int col  = wv * 16 + n_l;   // global output column for this lane's D elements

  __builtin_prefetch(Wbf + tid * 32, 0, 3);   // weights hot in L2

  // ---- gather neighbors (f32 + bf16 mirror, converted exactly once) ----
  {
    int m = tid >> 4, k = tid & 15;
    int nei = graph[(msg0 + m) * NEI + k];
    const float* src = h_in + nei * HID;
    float*          dstf = sh_hnei    + tid * LDH;
    unsigned short* dstb = sh_hnei_bf + tid * LDBH;
#pragma unroll
    for (int i = 0; i < 8; ++i) {
      float4 v = *(const float4*)(src + i * 4);
      *(float4*)(dstf + i * 4) = v;
      *(uint2*)(dstb + i * 4) = make_uint2(pk2(v.x, v.y), pk2(v.z, v.w));
    }
  }
  // ---- stage x tile as bf16 (coalesced, converted once) ----
  {
    int m = tid >> 4, c0 = (tid & 15) * 8;
    const float* xr = x + (msg0 + m) * HID + c0;
    float4 a = *(const float4*)(xr);
    float4 b = *(const float4*)(xr + 4);
    *(uint4*)(sh_x_bf + m * LDBH + c0) =
        make_uint4(pk2(a.x, a.y), pk2(a.z, a.w), pk2(b.x, b.y), pk2(b.z, b.w));
  }
  __syncthreads();

  // ---- sum_h over 16 neighbors (f32) + bf16 mirror ----
  {
    int m = tid >> 4, c0 = (tid & 15) * 8;
    float s[8];
#pragma unroll
    for (int c = 0; c < 8; ++c) {
      float acc = 0.f;
#pragma unroll
      for (int k = 0; k < NEI; ++k) acc += sh_hnei[(m * NEI + k) * LDH + c0 + c];
      s[c] = acc;
      sh_sumh[m * LDH + c0 + c] = acc;
    }
    *(uint4*)(sh_sumh_bf + m * LDBH + c0) =
        make_uint4(pk2(s[0], s[1]), pk2(s[2], s[3]), pk2(s[4], s[5]), pk2(s[6], s[7]));
  }

  // ---- r1 = x @ Wr^T for this tile (4 WMMA/wave) ----
  {
    v8f acc = {};
    const unsigned short* wrow = WrB + col * HID;
    const unsigned short* arow = sh_x_bf + n_l * LDBH;
#pragma unroll
    for (int kt = 0; kt < 4; ++kt)
      acc = wmma_bf16(load_afrag(arow, kt * 32 + half * 16),
                      load_bfrag(wrow, kt * 32 + half * 16), acc);
#pragma unroll
    for (int j = 0; j < 8; ++j)
      sh_r1[(j + half * 8) * LDH + col] = acc[j];
  }
  __syncthreads();

  // ---- z = sigmoid([x | sum_h] @ Wz^T + b) : 8 WMMA/wave, kept in registers ----
  v8f zacc = {};
  {
    const unsigned short* wrow = WzB + col * 256;
    const unsigned short* arow = sh_x_bf    + n_l * LDBH;
    const unsigned short* srow = sh_sumh_bf + n_l * LDBH;
#pragma unroll
    for (int kt = 0; kt < 4; ++kt)
      zacc = wmma_bf16(load_afrag(arow, kt * 32 + half * 16),
                       load_bfrag(wrow, kt * 32 + half * 16), zacc);
#pragma unroll
    for (int kt = 0; kt < 4; ++kt)
      zacc = wmma_bf16(load_afrag(srow, kt * 32 + half * 16),
                       load_bfrag(wrow, 128 + kt * 32 + half * 16), zacc);
    float bz = Wz_b[col];
#pragma unroll
    for (int j = 0; j < 8; ++j) zacc[j] = sigmoidf_(zacc[j] + bz);
  }

  // ---- r2 / gated sum: 16 neighbor row-tiles x 4 WMMA (dominant: 64 WMMA/wave) ----
  {
    const unsigned short* urow = UrB + col * HID;
    v16bf urb[4];
#pragma unroll
    for (int kt = 0; kt < 4; ++kt) urb[kt] = load_bfrag(urow, kt * 32 + half * 16);
    const float urbias = Ur_b[col];

    for (int rt = 0; rt < 16; ++rt) {       // rt = message index in tile
      const unsigned short* arow = sh_hnei_bf + (rt * NEI + n_l) * LDBH;
      v8f racc = {};
#pragma unroll
      for (int kt = 0; kt < 4; ++kt)
        racc = wmma_bf16(load_afrag(arow, kt * 32 + half * 16), urb[kt], racc);

      const float r1v = sh_r1[rt * LDH + col];   // r1[message rt, col] (broadcast)
      float part = 0.f;
#pragma unroll
      for (int j = 0; j < 8; ++j) {
        int M = j + half * 8;                    // neighbor index
        float rr = sigmoidf_(racc[j] + r1v + urbias);
        part += rr * sh_hnei[(rt * NEI + M) * LDH + col];   // f32 gating
      }
      part += __shfl_xor(part, 16, 32);          // merge the two M-halves
      if (lane < 16) sh_sumg_bf[rt * LDBH + col] = f2bf(part);
    }
  }
  __syncthreads();

  // ---- pre_h = tanh([x | sum_gated] @ Wh^T + b), then combine ----
  v8f hacc = {};
  {
    const unsigned short* wrow = WhB + col * 256;
    const unsigned short* arow = sh_x_bf    + n_l * LDBH;
    const unsigned short* grow = sh_sumg_bf + n_l * LDBH;
#pragma unroll
    for (int kt = 0; kt < 4; ++kt)
      hacc = wmma_bf16(load_afrag(arow, kt * 32 + half * 16),
                       load_bfrag(wrow, kt * 32 + half * 16), hacc);
#pragma unroll
    for (int kt = 0; kt < 4; ++kt)
      hacc = wmma_bf16(load_afrag(grow, kt * 32 + half * 16),
                       load_bfrag(wrow, 128 + kt * 32 + half * 16), hacc);
  }
  {
    const float bh = Wh_b[col];
#pragma unroll
    for (int j = 0; j < 8; ++j) {
      int M = j + half * 8;
      float ph = tanhf(hacc[j] + bh);
      float sh = sh_sumh[M * LDH + col];
      float z  = zacc[j];
      sh_r1[M * LDH + col] = (1.0f - z) * sh + z * ph;   // stage h_out (reuse r1 buf)
    }
  }
  __syncthreads();

  // ---- coalesced store; apply padding mask (global row 0 -> 0) ----
  {
    int m = tid >> 4, c0 = (tid & 15) * 8;
    int g = msg0 + m;
    float* dst = h_out + g * HID + c0;
    if (g == 0) {
#pragma unroll
      for (int i = 0; i < 8; ++i) dst[i] = 0.0f;
    } else {
#pragma unroll
      for (int i = 0; i < 8; ++i) dst[i] = sh_r1[m * LDH + c0 + i];
    }
  }
}

extern "C" void kernel_launch(void* const* d_in, const int* in_sizes, int n_in,
                              void* d_out, int out_size, void* d_ws, size_t ws_size,
                              hipStream_t stream) {
  const float* h     = (const float*)d_in[0];
  const float* x     = (const float*)d_in[1];
  const int*   graph = (const int*)  d_in[2];
  const float* Wz_w  = (const float*)d_in[3];
  const float* Wz_b  = (const float*)d_in[4];
  const float* Wr_w  = (const float*)d_in[5];
  const float* Ur_w  = (const float*)d_in[6];
  const float* Ur_b  = (const float*)d_in[7];
  const float* Wh_w  = (const float*)d_in[8];
  const float* Wh_b  = (const float*)d_in[9];
  float* out = (float*)d_out;

  const int N = in_sizes[0] / HID;   // 100000, divisible by 16

  unsigned short* wbf = (unsigned short*)d_ws;               // 196608 B of bf16 weights
  float* hping = (float*)((char*)d_ws + 196608);             // N*128 f32 ping buffer

  gru_cvt_weights<<<dim3(128), dim3(256), 0, stream>>>(Wz_w, Wr_w, Ur_w, Wh_w, wbf);

  const size_t shmem =
      (size_t)(256 * LDH + 2 * 16 * LDH) * sizeof(float) +
      (size_t)(256 * LDBH + 3 * 16 * LDBH) * sizeof(unsigned short);   // ~235 KB
  dim3 grid(N / 16), block(256);
  // depth 1..4, ping-ponging through ws buffer and d_out (used as scratch)
  gru_depth<<<grid, block, shmem, stream>>>(h,     x, graph, wbf, Wz_b, Ur_b, Wh_b, hping);
  gru_depth<<<grid, block, shmem, stream>>>(hping, x, graph, wbf, Wz_b, Ur_b, Wh_b, out);
  gru_depth<<<grid, block, shmem, stream>>>(out,   x, graph, wbf, Wz_b, Ur_b, Wh_b, hping);
  gru_depth<<<grid, block, shmem, stream>>>(hping, x, graph, wbf, Wz_b, Ur_b, Wh_b, out);
}
